// DirDeepModel_64768106824134
// MI455X (gfx1250) — matrix operand order repaired and checked by reading, not compiled, not run on hardware
//
#include <hip/hip_runtime.h>
#include <hip/hip_bf16.h>
#include <math.h>

// ---------------------------------------------------------------------------
// Model dims (fixed by the reference)
// ---------------------------------------------------------------------------
#define CNUM    128
#define NFACES  10000
#define NLAYERS 30
#define KDIM    256          // GEMM reduction depth (2*CNUM), all layers

typedef __attribute__((ext_vector_type(16))) __bf16          v16bf;
typedef __attribute__((ext_vector_type(8)))  float           v8f;
typedef __attribute__((ext_vector_type(4)))  float           v4f;
typedef __attribute__((ext_vector_type(4)))  unsigned int    v4u;
typedef __attribute__((ext_vector_type(8)))  unsigned int    v8u;

// fp32 -> bf16 round-to-nearest-even (bit pattern)
__device__ __forceinline__ unsigned int f2bf(float x) {
    unsigned int u = __float_as_uint(x);
    return (u + 0x7FFFu + ((u >> 16) & 1u)) >> 16;
}

__device__ __forceinline__ float elu1(float x) {
    return x > 0.f ? x : (expf(x) - 1.f);
}

// ---------------------------------------------------------------------------
// Elementwise / helper kernels
// ---------------------------------------------------------------------------
__global__ void fill_zero_k(float* __restrict__ p, int n) {
    int i = blockIdx.x * blockDim.x + threadIdx.x;
    if (i < n) p[i] = 0.f;
}

__global__ void elu_k(const float* __restrict__ x, float* __restrict__ y, int n) {
    int i = blockIdx.x * blockDim.x + threadIdx.x;
    if (i < n) y[i] = elu1(x[i]);
}

// Convert per-layer weight blocks [K,128] f32 -> transposed [128,K] bf16 (n-major)
__global__ void convert_bf16_T_k(const float* __restrict__ s, unsigned short* __restrict__ d, int n) {
    int i = blockIdx.x * blockDim.x + threadIdx.x;     // enumerates source (coalesced reads)
    if (i >= n) return;
    int layer = i / (KDIM * CNUM);
    int rem   = i - layer * (KDIM * CNUM);
    int k = rem >> 7;                // row in [K,128]
    int c = rem & (CNUM - 1);        // col
    d[(size_t)layer * (KDIM * CNUM) + (size_t)c * KDIM + k] = (unsigned short)f2bf(s[i]);
}

// v = inputs[N,3] @ W_in[3,128] + b_in
__global__ void input_proj_k(const float* __restrict__ in, const float* __restrict__ Win,
                             const float* __restrict__ bin, float* __restrict__ v, int rows) {
    int i = blockIdx.x * blockDim.x + threadIdx.x;
    if (i >= rows * CNUM) return;
    int r = i >> 7, c = i & (CNUM - 1);
    v[i] = in[r * 3 + 0] * Win[c] + in[r * 3 + 1] * Win[CNUM + c] +
           in[r * 3 + 2] * Win[2 * CNUM + c] + bin[c];
}

// ---------------------------------------------------------------------------
// COO SPMM: out[(4*NR),32] += vals[e] * x[cols[e],:] scattered to rows[e]
// one wave (32 lanes = 32 channels) per nonzero; native f32 atomics
// ---------------------------------------------------------------------------
__global__ void spmm_atomic_k(const int* __restrict__ rws, const int* __restrict__ cls,
                              const float* __restrict__ vls, const float* __restrict__ x,
                              float* __restrict__ out, int nnz) {
    int e    = blockIdx.x * (blockDim.x >> 5) + (threadIdx.x >> 5);
    int lane = threadIdx.x & 31;
    if (e < nnz) {
        float v = vls[e] * x[(size_t)cls[e] * 32 + lane];
        unsafeAtomicAdd(&out[(size_t)rws[e] * 32 + lane], v);
    }
}

// ---------------------------------------------------------------------------
// BatchNorm statistics: per-channel sum / sumsq over rows of X[rows,128]
// ---------------------------------------------------------------------------
__global__ void col_stats_k(const float* __restrict__ X, int rows,
                            float* __restrict__ sum, float* __restrict__ sumsq) {
    int c  = threadIdx.x;                // blockDim.x == 128
    int r0 = blockIdx.x * 64;
    int r1 = r0 + 64; if (r1 > rows) r1 = rows;
    float s = 0.f, s2 = 0.f;
    for (int r = r0; r < r1; ++r) {
        float v = X[(size_t)r * CNUM + c];
        s += v; s2 += v * v;
    }
    unsafeAtomicAdd(&sum[c], s);
    unsafeAtomicAdd(&sumsq[c], s2);
}

// masked per-channel sum + mask sum (for AvgResNet2 pooling)
__global__ void masked_colsum_k(const float* __restrict__ X, const float* __restrict__ mask,
                                int rows, float* __restrict__ sumx, float* __restrict__ msum) {
    int c  = threadIdx.x;                // 128
    int r0 = blockIdx.x * 64;
    int r1 = r0 + 64; if (r1 > rows) r1 = rows;
    float s = 0.f, m = 0.f;
    for (int r = r0; r < r1; ++r) {
        float mk = mask[r];
        s += X[(size_t)r * CNUM + c] * mk;
        if (c == 0) m += mk;
    }
    unsafeAtomicAdd(&sumx[c], s);
    if (c == 0) unsafeAtomicAdd(msum, m);
}

__global__ void broadcast_avg_k(float* __restrict__ xb, const float* __restrict__ sumx,
                                const float* __restrict__ msum, int n) {
    int i = blockIdx.x * blockDim.x + threadIdx.x;
    if (i < n) xb[i] = sumx[i & (CNUM - 1)] / msum[0];
}

// fold mean/var + gamma/beta into scale/shift:  BN(x) = x*scale + shift
__global__ void bn_finalize_k(const float* __restrict__ sum, const float* __restrict__ sumsq,
                              const float* __restrict__ gamma, const float* __restrict__ beta,
                              float* __restrict__ scale, float* __restrict__ shift, float inv_rows) {
    int c = threadIdx.x;                 // launched with nchan threads
    float mean = sum[c] * inv_rows;
    float var  = sumsq[c] * inv_rows - mean * mean;
    var = fmaxf(var, 0.f);
    float sc = gamma[c] * rsqrtf(var + 1e-5f);
    scale[c] = sc;
    shift[c] = beta[c] - mean * sc;
}

// ---------------------------------------------------------------------------
// Core: Y[rows,128] = BN([X0|X1]) @ W[256,128] + bias (+resid) (+ELU copy)
//   * W passed pre-transposed n-major bf16:  WbT[n][k], n<128, k<256
//   * 256 threads = 8 waves; block tile 128x128; wave tile 16x128
//   * K loop fully unrolled (8 steps) -> 64 static v_wmma_f32_16x16x32_bf16
//   * BN fused into A staging; fragments built with 16B LDS loads matching
//     the ISA 7.12.2 16-bit operand layouts (two contiguous 8-half runs/lane)
// ---------------------------------------------------------------------------
struct AB32 { v4u lo, hi; };   // 32B fragment container for bit_cast

__global__ __launch_bounds__(256)
void bn_gemm_wmma_k(const float* __restrict__ X0, const float* __restrict__ X1,
                    const float* __restrict__ scale, const float* __restrict__ shift,
                    const unsigned short* __restrict__ WbT, const float* __restrict__ bias,
                    const float* __restrict__ resid, float* __restrict__ out,
                    float* __restrict__ elu_out, int rows) {
    // rows padded to 40 halfwords (80B) so 16B chunks stay aligned
    __shared__ __align__(16) unsigned short ldsA [128][40];   // A: 128 rows x 32 K
    __shared__ __align__(16) unsigned short ldsBT[128][40];   // B^T: 128 n x 32 K

    const int tid  = threadIdx.x;
    const int wave = tid >> 5;
    const int lane = tid & 31;
    const int hi   = lane >> 4;          // 0 | 1
    const int lrow = lane & 15;
    const int blockRow = blockIdx.x * 128;

    // staging assignment: each thread owns 16 consecutive K of one row/col
    const int arow = tid >> 1;           // 0..127 (A row == B^T n-index)
    const int kseg = (tid & 1) * 16;     // 0 | 16

    int grow = blockRow + arow;          // clamp instead of branch: loads stay
    if (grow >= rows) grow = rows - 1;   // unconditional, stores are guarded
    const float* __restrict__ xrow0 = X0 + (size_t)grow * CNUM;
    const float* __restrict__ xrow1 = X1 + (size_t)grow * CNUM;
    const unsigned short* __restrict__ wrow = WbT + (size_t)arow * KDIM;

    v8f acc[8];
#pragma unroll
    for (int n = 0; n < 8; ++n)
#pragma unroll
        for (int j = 0; j < 8; ++j) acc[n][j] = 0.f;

#pragma unroll
    for (int kk = 0; kk < KDIM; kk += 32) {
        // ---- stage A tile: 16 floats -> BN -> bf16, two b128 LDS stores ----
        const float* __restrict__ xs = (kk < CNUM) ? xrow0 : xrow1;
        const v4f* xv4 = (const v4f*)(xs + (kk & (CNUM - 1)) + kseg);
        const v4f* sc4 = (const v4f*)(scale + kk + kseg);
        const v4f* sh4 = (const v4f*)(shift + kk + kseg);
        v4u packed[2];
#pragma unroll
        for (int q = 0; q < 4; ++q) {
            v4f x = xv4[q], s = sc4[q], h = sh4[q];
            unsigned int p0 = f2bf(x[0] * s[0] + h[0]) | (f2bf(x[1] * s[1] + h[1]) << 16);
            unsigned int p1 = f2bf(x[2] * s[2] + h[2]) | (f2bf(x[3] * s[3] + h[3]) << 16);
            packed[q >> 1][(q & 1) * 2 + 0] = p0;
            packed[q >> 1][(q & 1) * 2 + 1] = p1;
        }
        *(v4u*)&ldsA[arow][kseg + 0] = packed[0];
        *(v4u*)&ldsA[arow][kseg + 8] = packed[1];

        // ---- stage B tile: copy 16 consecutive bf16 of column-major weights ----
        const v4u* wv4 = (const v4u*)(wrow + kk + kseg);
        *(v4u*)&ldsBT[arow][kseg + 0] = wv4[0];
        *(v4u*)&ldsBT[arow][kseg + 8] = wv4[1];

        if (kk + 32 < KDIM)   // pull next weight chunk toward WGP (global_prefetch_b8)
            __builtin_prefetch(wrow + kk + 32 + kseg, 0, 1);
        __syncthreads();

        // ---- A fragment (ISA 16-bit 16x32 layout): K runs [k0,k0+8) & [k0+16,k0+24)
        const int ak0 = hi ? 8 : 0;
        AB32 af;
        af.lo = *(const v4u*)&ldsA[wave * 16 + lrow][ak0];
        af.hi = *(const v4u*)&ldsA[wave * 16 + lrow][ak0 + 16];
        v16bf a = __builtin_bit_cast(v16bf, af);

        // ---- 8 column tiles of B, 8 WMMAs ----
        const int bk0 = hi ? 8 : 0;      // lanes 0-15: K 0..7,16..23; 16-31: K 8..15,24..31
#pragma unroll
        for (int n = 0; n < 8; ++n) {
            AB32 bf;
            bf.lo = *(const v4u*)&ldsBT[n * 16 + lrow][bk0];
            bf.hi = *(const v4u*)&ldsBT[n * 16 + lrow][bk0 + 16];
            v16bf b = __builtin_bit_cast(v16bf, bf);
            acc[n] = __builtin_amdgcn_wmma_f32_16x16x32_bf16(
                false, a, false, b, (short)0, acc[n], false, false);
        }
        __syncthreads();
    }

    // ---- epilogue: bias (+resid) (+ELU mirror), C layout per ISA 7.12.2 ----
#pragma unroll
    for (int n = 0; n < 8; ++n) {
        int col = n * 16 + lrow;
        float bc = bias[col];
#pragma unroll
        for (int r = 0; r < 8; ++r) {
            int row = blockRow + wave * 16 + r + (hi << 3);
            if (row < rows) {
                size_t o = (size_t)row * CNUM + col;
                float val = acc[n][r] + bc;
                if (resid) val += resid[o];
                out[o] = val;
                if (elu_out) elu_out[o] = elu1(val);
            }
        }
    }
}

// ---------------------------------------------------------------------------
// Final head: out[r] = elu( (BN(v) @ W_out)[r] + b_out ), one wave per row
// ---------------------------------------------------------------------------
__global__ void bn_matvec_elu_k(const float* __restrict__ v, const float* __restrict__ scale,
                                const float* __restrict__ shift, const float* __restrict__ w,
                                const float* __restrict__ b, float* __restrict__ out, int rows) {
    int row  = blockIdx.x * (blockDim.x >> 5) + (threadIdx.x >> 5);
    int lane = threadIdx.x & 31;
    if (row >= rows) return;
    float s = 0.f;
#pragma unroll
    for (int j = 0; j < 4; ++j) {
        int c = lane + 32 * j;
        s += (v[(size_t)row * CNUM + c] * scale[c] + shift[c]) * w[c];
    }
    for (int o = 16; o > 0; o >>= 1) s += __shfl_down(s, o, 32);
    if (lane == 0) out[row] = elu1(s + b[0]);
}

// ---------------------------------------------------------------------------
// Host orchestration (graph-capture safe: kernels only, all on `stream`)
// ---------------------------------------------------------------------------
extern "C" void kernel_launch(void* const* d_in, const int* in_sizes, int n_in,
                              void* d_out, int out_size, void* d_ws, size_t ws_size,
                              hipStream_t stream) {
    (void)n_in; (void)out_size; (void)ws_size;

    const float* inputs  = (const float*)d_in[0];
    const float* mask    = (const float*)d_in[1];
    const int*   Di_r    = (const int*)  d_in[2];
    const int*   Di_c    = (const int*)  d_in[3];
    const float* Di_v    = (const float*)d_in[4];
    const int*   DiA_r   = (const int*)  d_in[5];
    const int*   DiA_c   = (const int*)  d_in[6];
    const float* DiA_v   = (const float*)d_in[7];
    const float* W_in    = (const float*)d_in[8];
    const float* b_in    = (const float*)d_in[9];
    const float* g0      = (const float*)d_in[10];
    const float* be0     = (const float*)d_in[11];
    const float* W0      = (const float*)d_in[12];
    const float* b0      = (const float*)d_in[13];
    const float* g1      = (const float*)d_in[14];
    const float* be1     = (const float*)d_in[15];
    const float* Wl1     = (const float*)d_in[16];
    const float* bl1     = (const float*)d_in[17];
    const float* g2      = (const float*)d_in[18];
    const float* be2     = (const float*)d_in[19];
    const float* W_out   = (const float*)d_in[20];
    const float* b_out   = (const float*)d_in[21];

    const int N   = in_sizes[0] / 3;     // 5000
    const int NF  = NFACES;              // 10000
    const int NNZ = in_sizes[2];         // 480000
    const int WSZ = KDIM * CNUM;         // weights per layer

    // ---- deterministic workspace carve-out ----
    char* wp = (char*)d_ws;
    auto carve = [&](size_t bytes) -> char* {
        char* p = wp; wp += (bytes + 255) & ~(size_t)255; return p;
    };
    float* v_buf = (float*)carve((size_t)N  * CNUM * 4);
    float* f_buf = (float*)carve((size_t)NF * CNUM * 4);
    float* xe    = (float*)carve((size_t)N  * CNUM * 4);   // elu(v)
    float* fe    = (float*)carve((size_t)NF * CNUM * 4);   // elu(f)
    float* Dv    = (float*)carve((size_t)NF * CNUM * 4);
    float* DAf   = (float*)carve((size_t)N  * CNUM * 4);
    float* feo   = (float*)carve((size_t)NF * CNUM * 4);   // elu(f_out)
    float* xb    = (float*)carve((size_t)N  * CNUM * 4);   // broadcast avg
    float* xcur  = (float*)carve((size_t)N  * CNUM * 4);
    float* sum   = (float*)carve(512 * 4);  float* sumsq = sum + 256;
    float* scale = (float*)carve(512 * 4);  float* shift = scale + 256;
    float* avgs  = (float*)carve(132 * 4);  float* msum  = avgs + 128;
    unsigned short* Wb0 = (unsigned short*)carve((size_t)NLAYERS * WSZ * 2);  // n-major bf16
    unsigned short* Wb1 = (unsigned short*)carve((size_t)NLAYERS * WSZ * 2);

    auto cdiv = [](int a, int b) { return (a + b - 1) / b; };

    auto zero = [&](float* p, int n) {
        fill_zero_k<<<cdiv(n, 256), 256, 0, stream>>>(p, n);
    };
    auto stats = [&](const float* X, int rows, float* s, float* s2) {
        col_stats_k<<<cdiv(rows, 64), 128, 0, stream>>>(X, rows, s, s2);
    };
    auto gemm = [&](const float* A0, const float* A1, const unsigned short* W,
                    const float* bias, const float* resid, float* out,
                    float* eout, int rows) {
        bn_gemm_wmma_k<<<cdiv(rows, 128), 256, 0, stream>>>(
            A0, A1, scale, shift, W, bias, resid, out, eout, rows);
    };

    // ---- setup: transposed bf16 weights, input projection, f = 0 ----
    convert_bf16_T_k<<<cdiv(NLAYERS * WSZ, 256), 256, 0, stream>>>(W0,  Wb0, NLAYERS * WSZ);
    convert_bf16_T_k<<<cdiv(NLAYERS * WSZ, 256), 256, 0, stream>>>(Wl1, Wb1, NLAYERS * WSZ);
    input_proj_k<<<cdiv(N * CNUM, 256), 256, 0, stream>>>(inputs, W_in, b_in, v_buf, N);
    zero(f_buf, NF * CNUM);

    for (int i = 0; i < NLAYERS; ++i) {
        const float* g0i  = g0  + (size_t)i * 256;  const float* be0i = be0 + (size_t)i * 256;
        const float* g1i  = g1  + (size_t)i * 256;  const float* be1i = be1 + (size_t)i * 256;
        const float* b0i  = b0  + (size_t)i * CNUM; const float* b1i  = bl1 + (size_t)i * CNUM;
        const unsigned short* W0i = Wb0 + (size_t)i * WSZ;
        const unsigned short* W1i = Wb1 + (size_t)i * WSZ;

        if ((i & 1) == 0) {
            // ---------- DirResNet2 ----------
            elu_k<<<cdiv(N * CNUM, 256), 256, 0, stream>>>(v_buf, xe, N * CNUM);
            elu_k<<<cdiv(NF * CNUM, 256), 256, 0, stream>>>(f_buf, fe, NF * CNUM);
            // Dv = Di @ xe  (quaternion-block view [4N,32] -> [4NF,32])
            zero(Dv, NF * CNUM);
            spmm_atomic_k<<<cdiv(NNZ, 8), 256, 0, stream>>>(Di_r, Di_c, Di_v, xe, Dv, NNZ);
            // f_out = BN([fe|Dv]) @ W0 + b0 ; also feo = elu(f_out)
            zero(sum, 512);
            stats(fe, NF, sum, sumsq);
            stats(Dv, NF, sum + 128, sumsq + 128);
            bn_finalize_k<<<1, 256, 0, stream>>>(sum, sumsq, g0i, be0i, scale, shift, 1.f / NF);
            gemm(fe, Dv, W0i, b0i, nullptr, f_buf, feo, NF);
            // DAf = DiA @ elu(f_out)
            zero(DAf, N * CNUM);
            spmm_atomic_k<<<cdiv(NNZ, 8), 256, 0, stream>>>(DiA_r, DiA_c, DiA_v, feo, DAf, NNZ);
            // v += BN([xe|DAf]) @ Wl1 + bl1
            zero(sum, 512);
            stats(xe, N, sum, sumsq);
            stats(DAf, N, sum + 128, sumsq + 128);
            bn_finalize_k<<<1, 256, 0, stream>>>(sum, sumsq, g1i, be1i, scale, shift, 1.f / N);
            gemm(xe, DAf, W1i, b1i, v_buf, v_buf, nullptr, N);
        } else {
            // ---------- AvgResNet2 ----------
            const float* src = v_buf;
            for (int sub = 0; sub < 2; ++sub) {
                elu_k<<<cdiv(N * CNUM, 256), 256, 0, stream>>>(src, xe, N * CNUM);
                zero(avgs, 129);
                masked_colsum_k<<<cdiv(N, 64), 128, 0, stream>>>(xe, mask, N, avgs, msum);
                broadcast_avg_k<<<cdiv(N * CNUM, 256), 256, 0, stream>>>(xb, avgs, msum, N * CNUM);
                zero(sum, 512);
                stats(xe, N, sum, sumsq);
                stats(xb, N, sum + 128, sumsq + 128);
                if (sub == 0) {
                    bn_finalize_k<<<1, 256, 0, stream>>>(sum, sumsq, g0i, be0i, scale, shift, 1.f / N);
                    gemm(xe, xb, W0i, b0i, nullptr, xcur, nullptr, N);
                    src = xcur;
                } else {
                    bn_finalize_k<<<1, 256, 0, stream>>>(sum, sumsq, g1i, be1i, scale, shift, 1.f / N);
                    gemm(xe, xb, W1i, b1i, v_buf, v_buf, nullptr, N);
                }
            }
        }
    }

    // ---- head: out = elu( BN(v) @ W_out + b_out ) ----
    zero(sum, 512);
    stats(v_buf, N, sum, sumsq);
    bn_finalize_k<<<1, 128, 0, stream>>>(sum, sumsq, g2, be2, scale, shift, 1.f / N);
    bn_matvec_elu_k<<<cdiv(N, 8), 256, 0, stream>>>(v_buf, scale, shift, W_out, b_out,
                                                    (float*)d_out, N);
}
